// CausalMachingIndicatorEstimator_46308337386020
// MI455X (gfx1250) — matrix operand order repaired and checked by reading, not compiled, not run on hardware
//
#include <hip/hip_runtime.h>
#include <hip/hip_bf16.h>

// ---------------------------------------------------------------------------
// MI455X (gfx1250) implementation.
//
// Roofline: ~60 GFLOP of dense GEMM vs ~200MB streamed activations (fits the
// 192MB L2) => compute-bound on the matrix path. All GEMMs are mapped onto
// v_wmma_f32_16x16x32_bf16 (bf16 A/B, fp32 accumulate). Each wave32 computes a
// 64x16 output tile (4 A-frags share 1 B-frag per k-step -> 4 WMMAs / k-step).
// N=5 graph algebra (aggregation, triangular solve) stays in fp32 VALU.
//
// Exact simplifications: (I - ac^T) is unit lower triangular -> forward
// substitution; trace((I+ac)^5) == N exactly -> avg_dag_loss == 0.
// ---------------------------------------------------------------------------

typedef __bf16 bf16_t;
typedef __attribute__((ext_vector_type(16))) __bf16 vbf16x16;
typedef __attribute__((ext_vector_type(8)))  float  vf32x8;

#define Z_ALPHA_F 1.959963984540054f

enum { EPI_BF16 = 0, EPI_RELU = 1, EPI_F32 = 2, EPI_SPSQ = 3,
       EPI_MISH = 4, EPI_LZ = 5, EPI_DEC2 = 6 };

__device__ __forceinline__ float softplusf(float x) {
    return (x > 20.f) ? x : log1pf(expf(x));
}
__device__ __forceinline__ float mishf(float x) {
    return x * tanhf(softplusf(x));
}

// ---------------------------------------------------------------------------
// Generic bf16 WMMA GEMM:  C[M x Nn] = epi( A[M x K] @ Bt[Nn x K]^T + bias )
// A row-major (row stride lda elements), Bt row-major (K contiguous per row).
// grid.z batches independent per-node weight sets (strides aZ/bZ/biasZ/cZ).
// ---------------------------------------------------------------------------
template<int EPI>
__global__ __launch_bounds__(128)
void wmma_gemm_k(const bf16_t* __restrict__ A, int lda, int aZ,
                 const bf16_t* __restrict__ Bt, int bZ,
                 const float* __restrict__ bias, int biasZ,
                 int M, int Nn, int K,
                 bf16_t* __restrict__ Cb, int ldc, int cZ,
                 float* __restrict__ Cf, int ldcf, int cfZ,
                 const float* __restrict__ cmp,
                 float* __restrict__ acc,
                 float* __restrict__ outMean,
                 float* __restrict__ outVar)
{
    __shared__ float red[128];
    const int tid   = threadIdx.x;
    const int wave  = tid >> 5;
    const int lane  = tid & 31;
    const int lhalf = lane >> 4;   // 0 or 1
    const int lmod  = lane & 15;
    const int z     = blockIdx.z;
    const int n0    = (blockIdx.y * 4 + wave) * 16;
    const long m0   = (long)blockIdx.x * 64;
    const bool active = (n0 < Nn) && (m0 < M);

    const bf16_t* Ab = A + (long)z * aZ;
    const bf16_t* Bb = Bt + (long)z * bZ;

    union FragU { uint4 q[2]; vbf16x16 v; };
    vf32x8 c[4] = {};

    if (active) {
        for (int k0 = 0; k0 < K; k0 += 32) {
            // B fragment: row n = n0+lmod, k = k0 + 16*lhalf .. +15 (contiguous)
            FragU bf_;
            const bf16_t* bp = Bb + (long)(n0 + lmod) * K + k0 + lhalf * 16;
            bf_.q[0] = *(const uint4*)(bp);
            bf_.q[1] = *(const uint4*)(bp + 8);
            #pragma unroll
            for (int t = 0; t < 4; ++t) {
                // A fragment: row m0+t*16+lmod, k = k0 + 16*lhalf .. +15 (contiguous)
                FragU af;
                const bf16_t* ap = Ab + (m0 + t * 16 + lmod) * (long)lda
                                      + k0 + lhalf * 16;
                af.q[0] = *(const uint4*)(ap);
                af.q[1] = *(const uint4*)(ap + 8);
                c[t] = __builtin_amdgcn_wmma_f32_16x16x32_bf16(
                          false, af.v, false, bf_.v, (short)0, c[t], false, false);
            }
        }
    }

    float lzpart = 0.f;
    if (active) {
        const int col = n0 + lmod;
        const float bv = bias ? bias[(long)z * biasZ + col] : 0.f;
        #pragma unroll
        for (int t = 0; t < 4; ++t) {
            #pragma unroll
            for (int r = 0; r < 8; ++r) {
                long row = m0 + t * 16 + lhalf * 8 + r;  // C layout: r + 8*(lane>>4)
                float v = c[t][r] + bv;
                if constexpr (EPI == EPI_RELU) v = fmaxf(v, 0.f);
                if constexpr (EPI == EPI_MISH) v = mishf(v);
                if constexpr (EPI == EPI_SPSQ) { float sp = softplusf(v); v = sp * sp; }
                if constexpr (EPI == EPI_BF16 || EPI == EPI_RELU || EPI == EPI_MISH) {
                    Cb[(long)z * cZ + row * (long)ldc + col] = (bf16_t)v;
                } else if constexpr (EPI == EPI_F32 || EPI == EPI_SPSQ) {
                    Cf[(long)z * cfZ + row * (long)ldcf + col] = v;
                } else if constexpr (EPI == EPI_LZ) {
                    float dd = v - cmp[row * (long)ldcf + col];
                    lzpart += dd * dd;
                } else if constexpr (EPI == EPI_DEC2) {
                    long bidx = (row * 5 + z) * 32;
                    if (col < 32) outMean[bidx + col] = v;
                    else          outVar[bidx + col - 32] = expf(v);
                }
            }
        }
    }

    if constexpr (EPI == EPI_LZ) {
        red[tid] = lzpart; __syncthreads();
        for (int s = 64; s > 0; s >>= 1) {
            if (tid < s) red[tid] += red[tid + s];
            __syncthreads();
        }
        if (tid == 0) atomicAdd(acc, red[0]);
    }
}

// ----------------------------- prep kernels --------------------------------
// out[z][n*K+k] = (bf16) in[z][k*N+n]   (transpose so WMMA B-rows are K-contig)
__global__ void tconv_k(const float* __restrict__ in, bf16_t* __restrict__ out,
                        int K, int N)
{
    int z = blockIdx.z;
    int idx = blockIdx.x * 256 + threadIdx.x;
    if (idx >= K * N) return;
    int n = idx / K, k = idx % K;
    out[(size_t)z * K * N + idx] = (bf16_t)in[(size_t)z * K * N + (size_t)k * N + n];
}

__global__ void cvt_k(const float* __restrict__ in, bf16_t* __restrict__ out, size_t n)
{
    size_t i = (size_t)blockIdx.x * 256 + threadIdx.x;
    size_t stride = (size_t)gridDim.x * 256;
    for (; i < n; i += stride) out[i] = (bf16_t)in[i];
}

// ACC layout (floats): 0 kl_raw, 1 ll, 2 se4, 3 bce, 4 lz,
// 5..36 colK, 37..68 colPIW, [uint] 69..100 colMinBits, 101..132 colMaxBits.
__global__ void init_acc_k(float* acc)
{
    int i = threadIdx.x;
    unsigned* ua = (unsigned*)acc;
    if (i < 69)       acc[i] = 0.f;
    else if (i < 101) ua[i] = 0x7f800000u;  // +inf
    else if (i < 256) ua[i] = 0u;
}

// ----------------------- GCN aggregation (N=5) -----------------------------
__global__ void agg_k(const bf16_t* __restrict__ g, const float* __restrict__ ew,
                      const int* __restrict__ esrc, const int* __restrict__ edst,
                      bf16_t* __restrict__ out)
{
    __shared__ float acm[25];
    int b = blockIdx.x, h = threadIdx.x;
    if (h < 25) acm[h] = 0.f;
    __syncthreads();
    if (h < 10) acm[esrc[h] * 5 + edst[h]] = ew[(size_t)b * 10 + h];
    __syncthreads();
    float gv[5];
    #pragma unroll
    for (int u = 0; u < 5; ++u) gv[u] = (float)g[((size_t)b * 5 + u) * 128 + h];
    #pragma unroll
    for (int v = 0; v < 5; ++v) {
        float a = gv[v];
        #pragma unroll
        for (int u = 0; u < 5; ++u) a += acm[u * 5 + v] * gv[u];
        out[((size_t)b * 5 + v) * 128 + h] = (bf16_t)a;
    }
}

// ------------------- KL partial + graph_feat (mean over n) -----------------
__global__ void klgf_k(const float* __restrict__ NM, const float* __restrict__ NV,
                       float* __restrict__ acc, bf16_t* __restrict__ gf)
{
    __shared__ float red[128];
    int b = blockIdx.x, h = threadIdx.x;
    float p = 0.f, s = 0.f;
    #pragma unroll
    for (int u = 0; u < 5; ++u) {
        float m = NM[((size_t)b * 5 + u) * 128 + h];
        float v = NV[((size_t)b * 5 + u) * 128 + h];
        p += m * m + v - logf(v + 1e-8f) - 1.f;
        s += m;
    }
    gf[(size_t)b * 128 + h] = (bf16_t)(s * 0.2f);
    red[h] = p; __syncthreads();
    for (int st = 64; st > 0; st >>= 1) {
        if (h < st) red[h] += red[h + st];
        __syncthreads();
    }
    if (h == 0) atomicAdd(&acc[0], red[0]);
}

// -------------------------- GRU cell ---------------------------------------
__global__ void gru_cell_k(const bf16_t* __restrict__ GIb, const bf16_t* __restrict__ GHb,
                           const float* __restrict__ hprev,
                           float* __restrict__ hn_out, bf16_t* __restrict__ hn_b)
{
    int b = blockIdx.x, h = threadIdx.x;
    size_t base = (size_t)b * 384;
    float ir = (float)GIb[base + h], iz = (float)GIb[base + 128 + h],
          in_ = (float)GIb[base + 256 + h];
    float hr = (float)GHb[base + h], hz = (float)GHb[base + 128 + h],
          hn2 = (float)GHb[base + 256 + h];
    float r  = 1.f / (1.f + expf(-(ir + hr)));
    float zz = 1.f / (1.f + expf(-(iz + hz)));
    float nn = tanhf(in_ + r * hn2);
    float hp = hprev[(size_t)b * 128 + h];
    float o  = (1.f - zz) * nn + zz * hp;
    hn_out[(size_t)b * 128 + h] = o;
    hn_b[(size_t)b * 128 + h] = (bf16_t)o;
}

// ---- causal solve (unit-lower-triangular forward subst.) + parent_sum ------
__global__ void causal_k(const float* __restrict__ NM, const float* __restrict__ ew,
                         const float* __restrict__ cm,
                         const int* __restrict__ esrc, const int* __restrict__ edst,
                         float* __restrict__ CZ, bf16_t* __restrict__ CZb,
                         bf16_t* __restrict__ PS)
{
    __shared__ float acm[25];
    __shared__ float pm[25];
    int b = blockIdx.x, h = threadIdx.x;
    if (h < 25) { acm[h] = 0.f; pm[h] = (cm[h] != 0.f) ? 1.f : 0.f; }
    __syncthreads();
    if (h < 10) {
        int s = esrc[h], d = edst[h];
        acm[s * 5 + d] = ew[(size_t)b * 10 + h] * cm[s * 5 + d];
    }
    __syncthreads();
    float zv[5];
    #pragma unroll
    for (int v = 0; v < 5; ++v) {
        float t = NM[((size_t)b * 5 + v) * 128 + h];
        #pragma unroll
        for (int u = 0; u < 5; ++u)
            if (u < v) t += acm[u * 5 + v] * zv[u];
        zv[v] = t;
        CZ[((size_t)b * 5 + v) * 128 + h] = t;
        CZb[((size_t)b * 5 + v) * 128 + h] = (bf16_t)t;
    }
    #pragma unroll
    for (int j = 0; j < 5; ++j) {
        float t = 0.f;
        #pragma unroll
        for (int i = 0; i < 5; ++i) t += pm[i * 5 + j] * zv[i];
        PS[((size_t)b * 5 + j) * 128 + h] = (bf16_t)t;
    }
}

// ----------------------- decoder-side loss reductions ----------------------
__global__ void loss_k(const float* __restrict__ yt_, const float* __restrict__ ym_,
                       const float* __restrict__ yv_, float* __restrict__ acc)
{
    __shared__ float red[256];
    __shared__ float ck[32], cp[32];
    __shared__ unsigned cmn[32], cmx[32];
    int tid = threadIdx.x;
    size_t r = (size_t)blockIdx.x * 8 + (tid >> 5);
    int d = tid & 31;
    size_t idx = r * 32 + d;
    float yt = yt_[idx], ym = ym_[idx], yv = yv_[idx];
    float vs = yv + 1e-6f;
    float diff = yt - ym;
    float ll = -0.5f * (diff * diff / vs + logf(6.283185307179586f * vs));
    int n = (int)(r % 5);
    float se  = (n < 4)  ? diff * diff : 0.f;
    float bce = (n == 4) ? (fmaxf(ym, 0.f) - ym * yt + log1pf(expf(-fabsf(ym)))) : 0.f;
    float s = sqrtf(yv + 1e-6f);
    float lo = ym - Z_ALPHA_F * s, up = ym + Z_ALPHA_F * s;
    float k = (yt >= lo && yt <= up) ? 1.f : 0.f;
    float piw = (up - lo) * k;

    if (tid < 32) { ck[tid] = 0.f; cp[tid] = 0.f; cmn[tid] = 0x7f800000u; cmx[tid] = 0u; }
    __syncthreads();
    atomicAdd(&ck[d], k);
    atomicAdd(&cp[d], piw);
    // y_true ~ U[0,1): non-negative => uint bit-pattern order == float order
    atomicMin(&cmn[d], __float_as_uint(yt));
    atomicMax(&cmx[d], __float_as_uint(yt));

    red[tid] = ll; __syncthreads();
    for (int st = 128; st > 0; st >>= 1) { if (tid < st) red[tid] += red[tid + st]; __syncthreads(); }
    if (tid == 0) atomicAdd(&acc[1], red[0]);
    __syncthreads();
    red[tid] = se; __syncthreads();
    for (int st = 128; st > 0; st >>= 1) { if (tid < st) red[tid] += red[tid + st]; __syncthreads(); }
    if (tid == 0) atomicAdd(&acc[2], red[0]);
    __syncthreads();
    red[tid] = bce; __syncthreads();
    for (int st = 128; st > 0; st >>= 1) { if (tid < st) red[tid] += red[tid + st]; __syncthreads(); }
    if (tid == 0) atomicAdd(&acc[3], red[0]);
    __syncthreads();

    if (tid < 32) {
        atomicAdd(&acc[5 + tid], ck[tid]);
        atomicAdd(&acc[37 + tid], cp[tid]);
        unsigned* ua = (unsigned*)acc;
        atomicMin(&ua[69 + tid], cmn[tid]);
        atomicMax(&ua[101 + tid], cmx[tid]);
    }
}

// ------------------------------- finalize ----------------------------------
__global__ void fin_k(const float* __restrict__ acc, float* __restrict__ out_tail)
{
    const float Bq = 32768.f;
    float kl = 0.5f * acc[0];
    float ll_mean = acc[1] / (Bq * 5.f * 32.f);
    float elbo = kl / Bq - ll_mean;
    float l_reg = (acc[2] / 32.f) / (4.f * Bq + 1e-6f);
    float l_cls = (acc[3] / 32.f) / (Bq + 1e-6f);
    float l_rec = l_reg + l_cls;
    const unsigned* ua = (const unsigned*)acc;
    float ksum = 0.f, pin = 0.f;
    for (int d = 0; d < 32; ++d) {
        float ckv = acc[5 + d];
        ksum += ckv;
        float ymin = __uint_as_float(ua[69 + d]);
        float ymax = __uint_as_float(ua[101 + d]);
        pin += acc[37 + d] / (ckv + 1e-6f) / (ymax - ymin + 1e-6f);
    }
    pin /= 32.f;
    float picp = ksum / (Bq * 5.f * 32.f);
    float l_pi = pin - sqrtf(5.f) * picp;
    float l_z = acc[4] / (Bq * 5.f * 128.f);
    float dag = 0.f;   // trace((I+ac)^5) == N exactly (strict upper-tri ac)
    float total = powf(elbo * l_rec * l_pi * l_z * (dag + 1e-6f), 0.2f);
    out_tail[0] = total;
    out_tail[1] = elbo;
    out_tail[2] = l_rec;
    out_tail[3] = l_pi;
    out_tail[4] = l_z;
    out_tail[5] = dag;
}

// ---------------------------------------------------------------------------
extern "C" void kernel_launch(void* const* d_in, const int* in_sizes, int n_in,
                              void* d_out_, int out_size, void* d_ws, size_t ws_size,
                              hipStream_t stream)
{
    (void)in_sizes; (void)n_in; (void)out_size; (void)ws_size;
    constexpr int B = 32768;
    const float* x       = (const float*)d_in[0];
    const float* y_true  = (const float*)d_in[1];
    const float* ew      = (const float*)d_in[2];
    const float* hist    = (const float*)d_in[3];
    const float* enc_W1  = (const float*)d_in[4];
    const float* enc_b1  = (const float*)d_in[5];
    const float* enc_W2  = (const float*)d_in[6];
    const float* enc_b2  = (const float*)d_in[7];
    const float* gcn_W   = (const float*)d_in[8];
    const float* gcn_b   = (const float*)d_in[9];
    const float* mu_W    = (const float*)d_in[10];
    const float* mu_b    = (const float*)d_in[11];
    const float* sig_W   = (const float*)d_in[12];
    const float* sig_b   = (const float*)d_in[13];
    const float* gru_Wih = (const float*)d_in[14];
    const float* gru_Whh = (const float*)d_in[15];
    const float* gru_bih = (const float*)d_in[16];
    const float* gru_bhh = (const float*)d_in[17];
    const float* cmat    = (const float*)d_in[18];
    const float* cm_W1   = (const float*)d_in[19];
    const float* cm_b1   = (const float*)d_in[20];
    const float* cm_W2   = (const float*)d_in[21];
    const float* cm_b2   = (const float*)d_in[22];
    const float* dec_W1  = (const float*)d_in[23];
    const float* dec_b1  = (const float*)d_in[24];
    const float* dec_W2  = (const float*)d_in[25];
    const float* dec_b2  = (const float*)d_in[26];
    const int*   esrc    = (const int*)d_in[27];
    const int*   edst    = (const int*)d_in[28];
    float* d_out = (float*)d_out_;

    const size_t VAR_OFF  = 5242880;      // y_pred_var
    const size_t GRU_OFF  = 10485760;     // new_gru_hidden (2,B,128)
    const size_t TAIL_OFF = 18874368;     // total_loss + losses[5]

    // ---- workspace carve (~340 MB total, 256B aligned) ----
    char* basep = (char*)d_ws;
    size_t off = 0;
    auto carve = [&](size_t bytes) -> void* {
        void* p = basep + off;
        off += (bytes + 255) & ~(size_t)255;
        return p;
    };
    bf16_t* encW1t = (bf16_t*)carve((size_t)5 * 128 * 32 * 2);
    bf16_t* encW2t = (bf16_t*)carve((size_t)5 * 128 * 128 * 2);
    bf16_t* gcnWt  = (bf16_t*)carve((size_t)2 * 128 * 128 * 2);
    bf16_t* muWt   = (bf16_t*)carve((size_t)128 * 128 * 2);
    bf16_t* sigWt  = (bf16_t*)carve((size_t)128 * 128 * 2);
    bf16_t* gWihb  = (bf16_t*)carve((size_t)2 * 384 * 128 * 2);
    bf16_t* gWhhb  = (bf16_t*)carve((size_t)2 * 384 * 128 * 2);
    bf16_t* cmW1t  = (bf16_t*)carve((size_t)128 * 128 * 2);
    bf16_t* cmW2t  = (bf16_t*)carve((size_t)128 * 128 * 2);
    bf16_t* decW1t = (bf16_t*)carve((size_t)5 * 128 * 128 * 2);
    bf16_t* decW2t = (bf16_t*)carve((size_t)5 * 64 * 128 * 2);
    bf16_t* xb     = (bf16_t*)carve((size_t)B * 5 * 32 * 2);
    bf16_t* histb  = (bf16_t*)carve((size_t)2 * B * 128 * 2);
    bf16_t* gf0    = (bf16_t*)carve((size_t)B * 128 * 2);
    bf16_t* gf1    = (bf16_t*)carve((size_t)B * 128 * 2);
    bf16_t* act0   = (bf16_t*)carve((size_t)B * 5 * 128 * 2);  // also GI_b / CZb
    bf16_t* act1   = (bf16_t*)carve((size_t)B * 5 * 128 * 2);  // also GH_b / PS / d1
    bf16_t* m1buf  = (bf16_t*)carve((size_t)B * 5 * 128 * 2);
    float*  NM     = (float*)carve((size_t)B * 5 * 128 * 4);
    float*  NVCZ   = (float*)carve((size_t)B * 5 * 128 * 4);   // node_var, then causal_z
    float*  acc    = (float*)carve(256 * 4);

    // ---- weight prep (tiny) ----
    tconv_k<<<dim3(16, 1, 5), 256, 0, stream>>>(enc_W1, encW1t, 32, 128);
    tconv_k<<<dim3(64, 1, 5), 256, 0, stream>>>(enc_W2, encW2t, 128, 128);
    tconv_k<<<dim3(64, 1, 2), 256, 0, stream>>>(gcn_W, gcnWt, 128, 128);
    tconv_k<<<dim3(64, 1, 1), 256, 0, stream>>>(mu_W, muWt, 128, 128);
    tconv_k<<<dim3(64, 1, 1), 256, 0, stream>>>(sig_W, sigWt, 128, 128);
    cvt_k<<<dim3(384, 1, 1), 256, 0, stream>>>(gru_Wih, gWihb, (size_t)2 * 384 * 128);
    cvt_k<<<dim3(384, 1, 1), 256, 0, stream>>>(gru_Whh, gWhhb, (size_t)2 * 384 * 128);
    tconv_k<<<dim3(64, 1, 1), 256, 0, stream>>>(cm_W1, cmW1t, 128, 128);
    tconv_k<<<dim3(64, 1, 1), 256, 0, stream>>>(cm_W2, cmW2t, 128, 128);
    tconv_k<<<dim3(64, 1, 5), 256, 0, stream>>>(dec_W1, decW1t, 128, 128);
    tconv_k<<<dim3(32, 1, 5), 256, 0, stream>>>(dec_W2, decW2t, 128, 64);
    cvt_k<<<dim3(20480, 1, 1), 256, 0, stream>>>(x, xb, (size_t)B * 5 * 32);
    cvt_k<<<dim3(32768, 1, 1), 256, 0, stream>>>(hist, histb, (size_t)2 * B * 128);
    init_acc_k<<<1, 256, 0, stream>>>(acc);

    // ---- encoder ----
    wmma_gemm_k<EPI_RELU><<<dim3(512, 2, 5), 128, 0, stream>>>(
        xb, 160, 32, encW1t, 4096, enc_b1, 128, B, 128, 32,
        act0, 640, 128, nullptr, 0, 0, nullptr, nullptr, nullptr, nullptr);
    wmma_gemm_k<EPI_BF16><<<dim3(512, 2, 5), 128, 0, stream>>>(
        act0, 640, 128, encW2t, 16384, enc_b2, 128, B, 128, 128,
        act1, 640, 128, nullptr, 0, 0, nullptr, nullptr, nullptr, nullptr);

    // ---- GCN x2 ----
    for (int l = 0; l < 2; ++l) {
        agg_k<<<dim3(B), 128, 0, stream>>>(act1, ew, esrc, edst, act0);
        wmma_gemm_k<EPI_RELU><<<dim3(2560, 2, 1), 128, 0, stream>>>(
            act0, 128, 0, gcnWt + (size_t)l * 16384, 0, gcn_b + l * 128, 0,
            B * 5, 128, 128, act1, 128, 0, nullptr, 0, 0,
            nullptr, nullptr, nullptr, nullptr);
    }

    // ---- mu / sigma heads ----
    wmma_gemm_k<EPI_F32><<<dim3(2560, 2, 1), 128, 0, stream>>>(
        act1, 128, 0, muWt, 0, mu_b, 0, B * 5, 128, 128,
        nullptr, 0, 0, NM, 128, 0, nullptr, nullptr, nullptr, nullptr);
    wmma_gemm_k<EPI_SPSQ><<<dim3(2560, 2, 1), 128, 0, stream>>>(
        act1, 128, 0, sigWt, 0, sig_b, 0, B * 5, 128, 128,
        nullptr, 0, 0, NVCZ, 128, 0, nullptr, nullptr, nullptr, nullptr);

    // ---- KL partial + graph_feat ----
    klgf_k<<<dim3(B), 128, 0, stream>>>(NM, NVCZ, acc, gf0);

    // ---- GRU x2 (gi/gh GEMMs reuse act0/act1 as bf16 preactivations) ----
    bf16_t* gin[2] = { gf0, gf1 };
    bf16_t* gout[2] = { gf1, gf0 };
    for (int l = 0; l < 2; ++l) {
        wmma_gemm_k<EPI_BF16><<<dim3(512, 6, 1), 128, 0, stream>>>(
            gin[l], 128, 0, gWihb + (size_t)l * 49152, 0, gru_bih + l * 384, 0,
            B, 384, 128, act0, 384, 0, nullptr, 0, 0,
            nullptr, nullptr, nullptr, nullptr);
        wmma_gemm_k<EPI_BF16><<<dim3(512, 6, 1), 128, 0, stream>>>(
            histb + (size_t)l * B * 128, 128, 0, gWhhb + (size_t)l * 49152, 0,
            gru_bhh + l * 384, 0, B, 384, 128, act1, 384, 0, nullptr, 0, 0,
            nullptr, nullptr, nullptr, nullptr);
        gru_cell_k<<<dim3(B), 128, 0, stream>>>(
            act0, act1, hist + (size_t)l * B * 128,
            d_out + GRU_OFF + (size_t)l * B * 128, gout[l]);
    }

    // ---- causal forward substitution + parent_sum (NVCZ becomes causal_z) ----
    causal_k<<<dim3(B), 128, 0, stream>>>(NM, ew, cmat, esrc, edst,
                                          NVCZ, act0 /*CZb*/, act1 /*PS*/);

    // ---- reconstruction MLP + l_z reduction ----
    wmma_gemm_k<EPI_MISH><<<dim3(2560, 2, 1), 128, 0, stream>>>(
        act1, 128, 0, cmW1t, 0, cm_b1, 0, B * 5, 128, 128,
        m1buf, 128, 0, nullptr, 0, 0, nullptr, nullptr, nullptr, nullptr);
    wmma_gemm_k<EPI_LZ><<<dim3(2560, 2, 1), 128, 0, stream>>>(
        m1buf, 128, 0, cmW2t, 0, cm_b2, 0, B * 5, 128, 128,
        nullptr, 0, 0, nullptr, 128, 0, NVCZ, acc + 4, nullptr, nullptr);

    // ---- decoder ----
    wmma_gemm_k<EPI_MISH><<<dim3(512, 2, 5), 128, 0, stream>>>(
        act0, 640, 128, decW1t, 16384, dec_b1, 128, B, 128, 128,
        act1, 640, 128, nullptr, 0, 0, nullptr, nullptr, nullptr, nullptr);
    wmma_gemm_k<EPI_DEC2><<<dim3(512, 1, 5), 128, 0, stream>>>(
        act1, 640, 128, decW2t, 8192, dec_b2, 64, B, 64, 128,
        nullptr, 0, 0, nullptr, 0, 0, nullptr, nullptr,
        d_out, d_out + VAR_OFF);

    // ---- losses ----
    loss_k<<<dim3(20480), 256, 0, stream>>>(y_true, d_out, d_out + VAR_OFF, acc);
    fin_k<<<1, 1, 0, stream>>>(acc, d_out + TAIL_OFF);
}
